// VQCodebook_83227876262438
// MI455X (gfx1250) — matrix operand order repaired and checked by reading, not compiled, not run on hardware
//
#include <hip/hip_runtime.h>
#include <hip/hip_bf16.h>
#include <float.h>

#define N_TOK   32768
#define K_CODE  8192
#define D_DIM   512
#define DECAYF  0.99f
#define COMMITF 0.25f
#define EPSF    1e-5f

typedef __attribute__((ext_vector_type(16))) __bf16        v16bf;
typedef __attribute__((ext_vector_type(8)))  float         v8f;
typedef __attribute__((ext_vector_type(4)))  unsigned int  v4u;

#define WMMA_BF16(A, B, C) \
  __builtin_amdgcn_wmma_f32_16x16x32_bf16(false, (A), false, (B), (short)0, (C), false, false)

// ---------------- workspace layout (bytes) ----------------
#define WS_IDX    ((size_t)0)                         // int32 [N]
#define WS_CLUST  ((size_t)131072)                    // f32 [K]
#define WS_LOSS   ((size_t)163840)                    // f32 [1]
#define WS_NTOT   ((size_t)163856)                    // f32 [1]
#define WS_ENORM  ((size_t)196608)                    // f32 [K]
#define WS_EHI    ((size_t)262144)                    // bf16 [K*D]
#define WS_ELO    (WS_EHI + (size_t)K_CODE*D_DIM*2)   // bf16 [K*D]
#define WS_ESUM   (WS_ELO + (size_t)K_CODE*D_DIM*2)   // f32 [K*D]

// ---------------- d_out layout (float elements) ----------------
#define O_ZQ     ((size_t)0)
#define O_IDX    ((size_t)N_TOK * D_DIM)
#define O_LOSS   (O_IDX + N_TOK)
#define O_NEMB   (O_LOSS + 1)
#define O_NCS    (O_NEMB + (size_t)K_CODE * D_DIM)
#define O_NES    (O_NCS + K_CODE)
#define O_USAGE  (O_NES + (size_t)K_CODE * D_DIM)

// ---------------- LDS layout for main kernel (bytes) ----------------
// bf16 rows padded to 1040B (520 halves): 16B-aligned, 4-bank shift/row -> conflict free
#define ROWB   1040
#define L_ZHI  0
#define L_ZLO  (L_ZHI + 128 * ROWB)
#define L_EHI  (L_ZLO + 128 * ROWB)
#define L_ELO  (L_EHI + 16 * ROWB)
#define L_TOT  (L_ELO + 16 * ROWB)   // 299,520 B  (<= 320 KB WGP LDS)

// ============================================================
// 0) zero accumulators: embed_sum[K*D], cluster[K], loss[1]
// ============================================================
__global__ void vq_zero_kernel(char* ws) {
  size_t i = (size_t)blockIdx.x * blockDim.x + threadIdx.x;
  const size_t tot = (size_t)K_CODE * D_DIM + K_CODE + 1;
  if (i >= tot) return;
  if (i < (size_t)K_CODE * D_DIM) {
    ((float*)(ws + WS_ESUM))[i] = 0.0f;
  } else if (i < (size_t)K_CODE * D_DIM + K_CODE) {
    ((float*)(ws + WS_CLUST))[i - (size_t)K_CODE * D_DIM] = 0.0f;
  } else {
    ((float*)(ws + WS_LOSS))[0] = 0.0f;
  }
}

// ============================================================
// 1) split embeddings into bf16 hi/lo + row norms
//    grid: K blocks x 256 threads
// ============================================================
__global__ void vq_prep_emb_kernel(const float* __restrict__ emb, char* ws) {
  __shared__ float red[256];
  const int k = blockIdx.x;
  const int t = threadIdx.x;
  __bf16* ehi = (__bf16*)(ws + WS_EHI);
  __bf16* elo = (__bf16*)(ws + WS_ELO);
  float ss = 0.0f;
#pragma unroll
  for (int i = 0; i < D_DIM / 256; ++i) {
    int d = t + i * 256;
    float x = emb[(size_t)k * D_DIM + d];
    __bf16 h = (__bf16)x;
    __bf16 l = (__bf16)(x - (float)h);
    ehi[(size_t)k * D_DIM + d] = h;
    elo[(size_t)k * D_DIM + d] = l;
    ss += x * x;
  }
  red[t] = ss;
  __syncthreads();
  for (int s = 128; s > 0; s >>= 1) {
    if (t < s) red[t] += red[t + s];
    __syncthreads();
  }
  if (t == 0) ((float*)(ws + WS_ENORM))[k] = red[0];
}

// ============================================================
// 2) fused split-bf16 WMMA GEMM + argmin over K
//    grid: N/128 blocks x 256 threads (8 waves, 16 rows/wave)
// ============================================================
__global__ void vq_argmin_kernel(const float* __restrict__ z, const char* __restrict__ ws,
                                 float* __restrict__ out, int* __restrict__ idx_ws) {
  extern __shared__ char smem[];
  const int t    = threadIdx.x;
  const int lane = t & 31;
  const int wave = t >> 5;          // 0..7
  const int l15  = lane & 15;
  const int sel  = lane >> 4;       // 0 / 1
  const int m0   = wave * 16;       // row tile inside block
  const int base = blockIdx.x * 128;

  const __bf16* ehi_g = (const __bf16*)(ws + WS_EHI);
  const __bf16* elo_g = (const __bf16*)(ws + WS_ELO);
  const float*  enorm = (const float*)(ws + WS_ENORM);

  // ---- stage 128 z rows as bf16 hi/lo in LDS ----
  for (int i = t; i < 128 * D_DIM; i += 256) {
    int r = i >> 9, c = i & (D_DIM - 1);
    float x = z[(size_t)(base + r) * D_DIM + c];
    __bf16 h = (__bf16)x;
    __bf16 l = (__bf16)(x - (float)h);
    *(__bf16*)(smem + L_ZHI + r * ROWB + c * 2) = h;
    *(__bf16*)(smem + L_ZLO + r * ROWB + c * 2) = l;
  }

  float minv[8];
  int   mini[8];
#pragma unroll
  for (int v = 0; v < 8; ++v) { minv[v] = FLT_MAX; mini[v] = 0; }

  const int NCHUNK = K_CODE / 16;   // 512
  for (int ct = 0; ct < NCHUNK; ++ct) {
    __syncthreads();
    // ---- cooperative stage of 16-code E tile (hi/lo) ----
    {
      const unsigned int* sh = (const unsigned int*)(ehi_g + (size_t)ct * 16 * D_DIM);
      const unsigned int* sl = (const unsigned int*)(elo_g + (size_t)ct * 16 * D_DIM);
#pragma unroll
      for (int j = 0; j < 16; ++j) {               // 16 rows * 256 uints = 4096
        int u  = t + j * 256;
        int r  = u >> 8, c2 = u & 255;
        *(unsigned int*)(smem + L_EHI + r * ROWB + c2 * 4) = sh[u];
        *(unsigned int*)(smem + L_ELO + r * ROWB + c2 * 4) = sl[u];
      }
      if (ct + 1 < NCHUNK) {                       // global_prefetch_b8 next tile
        __builtin_prefetch((const char*)(ehi_g + (size_t)(ct + 1) * 16 * D_DIM) + t * 64, 0, 1);
        __builtin_prefetch((const char*)(elo_g + (size_t)(ct + 1) * 16 * D_DIM) + t * 64, 0, 1);
      }
    }
    __syncthreads();

    v8f acc = {};
    const int arow = (m0 + l15) * ROWB;
    const int brow = l15 * ROWB;
#pragma unroll 4
    for (int kt = 0; kt < D_DIM / 32; ++kt) {      // 16 k-steps
      union { v16bf v; v4u u[2]; } ah, al, bh, bl;
      // A frag (16x32 bf16): V0-3 = K[sel*8 .. +8), V4-7 = K[16+sel*8 .. +8)
      int ao = arow + kt * 64 + sel * 16;
      ah.u[0] = *(const v4u*)(smem + L_ZHI + ao);
      ah.u[1] = *(const v4u*)(smem + L_ZHI + ao + 32);
      al.u[0] = *(const v4u*)(smem + L_ZLO + ao);
      al.u[1] = *(const v4u*)(smem + L_ZLO + ao + 32);
      // B frag (32x16 bf16): V0-3 = K[sel*8 .. +8), V4-7 = K[16+sel*8 .. +8)
      int bo = brow + kt * 64 + sel * 16;
      bh.u[0] = *(const v4u*)(smem + L_EHI + bo);
      bh.u[1] = *(const v4u*)(smem + L_EHI + bo + 32);
      bl.u[0] = *(const v4u*)(smem + L_ELO + bo);
      bl.u[1] = *(const v4u*)(smem + L_ELO + bo + 32);

      acc = WMMA_BF16(ah.v, bh.v, acc);            // zh*eh
      acc = WMMA_BF16(ah.v, bl.v, acc);            // zh*el
      acc = WMMA_BF16(al.v, bh.v, acc);            // zl*eh
    }

    // dist = ||e||^2 - 2 z.e   (||z||^2 constant per row -> dropped)
    const int nidx = ct * 16 + l15;
    const float en = enorm[nidx];
#pragma unroll
    for (int v = 0; v < 8; ++v) {
      float d = en - 2.0f * acc[v];
      if (d < minv[v]) { minv[v] = d; mini[v] = nidx; }
    }
  }

  // ---- cross-lane argmin within each 16-lane half ----
#pragma unroll
  for (int m = 1; m < 16; m <<= 1) {
#pragma unroll
    for (int v = 0; v < 8; ++v) {
      float ov = __shfl_xor(minv[v], m, 32);
      int   oi = __shfl_xor(mini[v], m, 32);
      if (ov < minv[v] || (ov == minv[v] && oi < mini[v])) { minv[v] = ov; mini[v] = oi; }
    }
  }
  if (l15 == 0) {
#pragma unroll
    for (int v = 0; v < 8; ++v) {
      int row = base + m0 + v + sel * 8;   // C layout: lanes<16 -> M=v, lanes>=16 -> M=v+8
      idx_ws[row] = mini[v];
      out[O_IDX + row] = (float)mini[v];
    }
  }
}

// ============================================================
// 3) gather z_q, commitment-loss partial, segment sums (atomics)
//    grid: N blocks x 256 threads
// ============================================================
__global__ void vq_scatter_kernel(const float* __restrict__ z, const float* __restrict__ emb,
                                  char* ws, float* __restrict__ out) {
  __shared__ float red[256];
  const int n = blockIdx.x;
  const int t = threadIdx.x;
  const int idx = ((const int*)(ws + WS_IDX))[n];
  float* esum  = (float*)(ws + WS_ESUM);
  float ls = 0.0f;
#pragma unroll
  for (int i = 0; i < D_DIM / 256; ++i) {
    int d = t + i * 256;
    float zq = emb[(size_t)idx * D_DIM + d];
    float zv = z[(size_t)n * D_DIM + d];
    out[O_ZQ + (size_t)n * D_DIM + d] = zq;       // straight-through value == z_q
    atomicAdd(&esum[(size_t)idx * D_DIM + d], zv);
    float df = zq - zv;
    ls += df * df;
  }
  red[t] = ls;
  __syncthreads();
  for (int s = 128; s > 0; s >>= 1) {
    if (t < s) red[t] += red[t + s];
    __syncthreads();
  }
  if (t == 0) {
    atomicAdd((float*)(ws + WS_LOSS), red[0]);
    atomicAdd((float*)(ws + WS_CLUST) + idx, 1.0f);
  }
}

// ============================================================
// 4) EMA cluster size + usage + n_total reduction (1 block)
// ============================================================
__global__ void vq_ema_cs_kernel(const float* __restrict__ ema_cs,
                                 const float* __restrict__ usage,
                                 char* ws, float* __restrict__ out) {
  __shared__ float red[1024];
  const int t = threadIdx.x;
  const float* clust = (const float*)(ws + WS_CLUST);
  float s = 0.0f;
  for (int k = t; k < K_CODE; k += 1024) {
    float c = clust[k];
    float ncs = DECAYF * ema_cs[k] + (1.0f - DECAYF) * c;
    out[O_NCS + k] = ncs;
    out[O_USAGE + k] = usage[k] + c;
    s += ncs;
  }
  red[t] = s;
  __syncthreads();
  for (int st = 512; st > 0; st >>= 1) {
    if (t < st) red[t] += red[t + st];
    __syncthreads();
  }
  if (t == 0) ((float*)(ws + WS_NTOT))[0] = red[0];
}

// ============================================================
// 5) EMA embed sum, normalized embeddings, loss scalar
// ============================================================
__global__ void vq_finalize_kernel(const float* __restrict__ ema_es,
                                   const char* __restrict__ ws, float* __restrict__ out) {
  size_t i = (size_t)blockIdx.x * blockDim.x + threadIdx.x;
  if (i >= (size_t)K_CODE * D_DIM) return;
  int k = (int)(i >> 9);
  float es  = ((const float*)(ws + WS_ESUM))[i];
  float nes = DECAYF * ema_es[i] + (1.0f - DECAYF) * es;
  out[O_NES + i] = nes;
  float ncs  = out[O_NCS + k];
  float ntot = ((const float*)(ws + WS_NTOT))[0];
  float sm   = (ncs + EPSF) / (ntot + (float)K_CODE * EPSF) * ntot;
  out[O_NEMB + i] = nes / sm;
  if (i == 0) {
    float la = ((const float*)(ws + WS_LOSS))[0];
    out[O_LOSS] = COMMITF * la / (float)((size_t)N_TOK * D_DIM);
  }
}

// ============================================================
extern "C" void kernel_launch(void* const* d_in, const int* in_sizes, int n_in,
                              void* d_out, int out_size, void* d_ws, size_t ws_size,
                              hipStream_t stream) {
  const float* z      = (const float*)d_in[0];
  const float* emb    = (const float*)d_in[1];
  const float* ema_cs = (const float*)d_in[2];
  const float* ema_es = (const float*)d_in[3];
  const float* usage  = (const float*)d_in[4];
  float* out = (float*)d_out;
  char*  ws  = (char*)d_ws;

  {
    size_t tot = (size_t)K_CODE * D_DIM + K_CODE + 1;
    vq_zero_kernel<<<dim3((unsigned)((tot + 255) / 256)), dim3(256), 0, stream>>>(ws);
  }
  vq_prep_emb_kernel<<<dim3(K_CODE), dim3(256), 0, stream>>>(emb, ws);
  vq_argmin_kernel<<<dim3(N_TOK / 128), dim3(256), L_TOT, stream>>>(
      z, ws, out, (int*)(ws + WS_IDX));
  vq_scatter_kernel<<<dim3(N_TOK), dim3(256), 0, stream>>>(z, emb, ws, out);
  vq_ema_cs_kernel<<<dim3(1), dim3(1024), 0, stream>>>(ema_cs, usage, ws, out);
  vq_finalize_kernel<<<dim3((unsigned)(((size_t)K_CODE * D_DIM + 255) / 256)), dim3(256), 0,
                       stream>>>(ema_es, ws, out);
}
